// KanLSTM_73117523247737
// MI455X (gfx1250) — compile-verified, matrix operand order checked by reading, not run on hardware
//
#include <hip/hip_runtime.h>
#include <cstdint>

// ---------------------------------------------------------------------------
// KAN-LSTM for MI455X (gfx1250, wave32, WMMA + Tensor Data Mover).
//
//   Phase 1 (parallel):   sx[t] = kan2(x_t) for all 16384 tokens
//   Phase 2 (sequential): per-step h-path kan2(h) + fused LSTM gate epilogue
//   Phase 3 (parallel):   out = kan_layer(ys)
//
// GEMMs: D = basis@ws + silu@wb + bias (dual-K fp32 WMMA accumulation).
// A/B tiles are staged into LDS with TENSOR_LOAD_TO_LDS (TDM, TENSORcnt),
// double-buffered and overlapped with v_wmma_f32_16x16x32_bf16 compute.
// ---------------------------------------------------------------------------

typedef __attribute__((ext_vector_type(16))) __bf16   v16bf;
typedef __attribute__((ext_vector_type(8)))  float    v8f;
typedef __attribute__((ext_vector_type(4)))  unsigned u32x4;
typedef __attribute__((ext_vector_type(8)))  int      i32x8;
typedef __attribute__((ext_vector_type(4)))  int      i32x4;

union FragBF {
    v16bf v;
    u32x4 q[2];
};

#if defined(__gfx1250__) && __has_builtin(__builtin_amdgcn_tensor_load_to_lds) && \
    __has_builtin(__builtin_amdgcn_s_wait_tensorcnt)
#define USE_TDM 1
#else
#define USE_TDM 0
#endif

// ------------------------- LN + RBF-basis + silu ---------------------------
__global__ __launch_bounds__(256) void kan_prep(const float* __restrict__ X, long ldx,
                                                int rows, int D,
                                                __bf16* __restrict__ basis,
                                                __bf16* __restrict__ silu) {
    int wid  = blockIdx.x * 8 + (threadIdx.x >> 5);
    int lane = threadIdx.x & 31;
    if (wid >= rows) return;
    const float* xr = X + (long)wid * ldx;

    float s = 0.f, s2 = 0.f;
    for (int d = lane; d < D; d += 32) { float v = xr[d]; s += v; s2 += v * v; }
    for (int o = 16; o > 0; o >>= 1) {
        s  += __shfl_xor(s,  o, 32);
        s2 += __shfl_xor(s2, o, 32);
    }
    float inv  = 1.f / (float)D;
    float mu   = s * inv;
    float var  = fmaxf(s2 * inv - mu * mu, 0.f);
    float rstd = rsqrtf(var + 1e-5f);

    __bf16* br = basis + (long)wid * (long)(D * 5);
    __bf16* sr = silu  + (long)wid * (long)D;
    for (int d = lane; d < D; d += 32) {
        float v  = xr[d];
        float xn = (v - mu) * rstd;
#pragma unroll
        for (int g = 0; g < 5; ++g) {       // GRID = {-2,-1,0,1,2}, inv_denom = 1
            float t = xn - (float)(g - 2);
            br[d * 5 + g] = (__bf16)__expf(-t * t);
        }
        sr[d] = (__bf16)(v / (1.f + __expf(-v)));
    }
}

// ---------------- TDM: 2D bf16 tile (rows x ktElems) global -> LDS ---------
// D# per CDNA5 ISA ch.8: group0 = {count/lds_addr/global_addr/type},
// group1 = {data_size=4B, tensor dims (dwords), tile dims, dim0 stride}.
__device__ __forceinline__ void tdm_load_2d(const __bf16* gbase, long rowStrideElems,
                                            int rows, int ktElems, unsigned ldsByteOff) {
#if USE_TDM
    unsigned long long ga = (unsigned long long)(uintptr_t)gbase;
    unsigned td0 = (unsigned)(ktElems >> 1);          // dwords per tile row
    unsigned td1 = (unsigned)rows;
    unsigned st0 = (unsigned)(rowStrideElems >> 1);   // row pitch in dwords

    u32x4 g0;
    g0[0] = 1u;                                        // count=1 valid descriptor
    g0[1] = ldsByteOff;                                // LDS byte address
    g0[2] = (unsigned)ga;                              // global_addr[31:0]
    g0[3] = (unsigned)((ga >> 32) & 0x01FFFFFFu) | (2u << 30);  // [56:32] | type=2

    i32x8 g1;
    g1[0] = (int)(2u << 16);                           // data_size = 4 bytes
    g1[1] = (int)((td0 & 0xFFFFu) << 16);              // tensor_dim0[15:0]
    g1[2] = (int)((td0 >> 16) | ((td1 & 0xFFFFu) << 16)); // dim0 hi | dim1 lo
    g1[3] = (int)((td1 >> 16) | (td0 << 16));          // dim1 hi | tile_dim0
    g1[4] = (int)(td1 & 0xFFFFu);                      // tile_dim1 (tile_dim2=0)
    g1[5] = (int)st0;                                  // tensor_dim0_stride lo
    g1[6] = 0;                                         // stride hi | dim1_stride lo
    g1[7] = 0;

    i32x4 z4 = {0, 0, 0, 0};
#if __clang_major__ >= 23
    i32x8 z8 = {0, 0, 0, 0, 0, 0, 0, 0};
    __builtin_amdgcn_tensor_load_to_lds(g0, g1, z4, z4, z8, 0);
#else
    __builtin_amdgcn_tensor_load_to_lds(g0, g1, z4, z4, 0);
#endif
#else
    (void)gbase; (void)rowStrideElems; (void)rows; (void)ktElems; (void)ldsByteOff;
#endif
}

// Fallback cooperative copy (all 256 threads), when TDM builtins are absent.
__device__ __forceinline__ void stage_copy(__bf16* dst, const __bf16* g, long ld,
                                           int rows, int kt) {
    int chunksPerRow = kt >> 3;                 // u32x4 = 8 bf16
    int total = rows * chunksPerRow;
    for (int i = threadIdx.x; i < total; i += 256) {
        int r = i / chunksPerRow, c = i - r * chunksPerRow;
        *(u32x4*)(dst + r * kt + (c << 3)) = *(const u32x4*)(g + (long)r * ld + (c << 3));
    }
}

// ----------------------------- dual-K WMMA GEMM ----------------------------
// Block: 128-row M strip x 16 N cols. 8 waves, one 16x16 tile each.
// K is processed in KT=64 tiles, TDM double-buffered through LDS.
// mode 0: C = acc + bias.  mode 1: fused LSTM gate update.
__global__ __launch_bounds__(256) void kan_gemm(
        const __bf16* __restrict__ A1, const __bf16* __restrict__ W1, int K1,
        const __bf16* __restrict__ A2, const __bf16* __restrict__ W2, int K2,
        const float* __restrict__ bias,
        float* __restrict__ C, long ldc,
        int M, int N, int mode,
        const float* __restrict__ sx_t, long ld_sx,
        float* __restrict__ cstate, float* __restrict__ hstate,
        float* __restrict__ ys_t, long ld_ys) {
    constexpr int KT = 64;
    __shared__ __bf16 sA[2][128 * KT];   // 2 x 16 KB
    __shared__ __bf16 sB[2][16 * KT];    // 2 x  2 KB

    int ntiles = N >> 4;
    int mtiles = M >> 4;
    int mb   = blockIdx.x / ntiles;      // 128-row strip index
    int tn   = blockIdx.x % ntiles;
    int w    = threadIdx.x >> 5;
    int lane = threadIdx.x & 31;
    int half = lane >> 4;
    int l15  = lane & 15;
    int ho   = half * 8;                 // K sub-offset (16-bit A/B VGPR layout)
    int tm   = mb * 8 + w;
    bool active = (tm < mtiles);
    int rstrip  = min(128, M - mb * 128);
    bool issuer = (threadIdx.x < 32);

    v8f acc = {0.f, 0.f, 0.f, 0.f, 0.f, 0.f, 0.f, 0.f};

#pragma unroll
    for (int ph = 0; ph < 2; ++ph) {
        const __bf16* A = ph ? A2 : A1;
        const __bf16* W = ph ? W2 : W1;
        int K = ph ? K2 : K1;
        int nt = K / KT;
        const __bf16* aStrip = A + (long)mb * 128 * K;
        const __bf16* bStrip = W + (long)tn * 16 * K;

        // preload tile 0 into buffer 0
#if USE_TDM
        if (issuer) {
            tdm_load_2d(aStrip, K, rstrip, KT, (unsigned)(uintptr_t)&sA[0][0]);
            tdm_load_2d(bStrip, K, 16,     KT, (unsigned)(uintptr_t)&sB[0][0]);
        }
#else
        stage_copy(&sA[0][0], aStrip, K, rstrip, KT);
        stage_copy(&sB[0][0], bStrip, K, 16,     KT);
#endif
        for (int it = 0; it < nt; ++it) {
            int cur = it & 1;
            if (it + 1 < nt) {   // prefetch next K tile into the other buffer
#if USE_TDM
                if (issuer) {
                    tdm_load_2d(aStrip + (it + 1) * KT, K, rstrip, KT,
                                (unsigned)(uintptr_t)&sA[cur ^ 1][0]);
                    tdm_load_2d(bStrip + (it + 1) * KT, K, 16, KT,
                                (unsigned)(uintptr_t)&sB[cur ^ 1][0]);
                }
#else
                stage_copy(&sA[cur ^ 1][0], aStrip + (it + 1) * KT, K, rstrip, KT);
                stage_copy(&sB[cur ^ 1][0], bStrip + (it + 1) * KT, K, 16, KT);
#endif
            }
#if USE_TDM
            // retire the 2 TDM ops of the *current* buffer (the 2 just issued
            // for the next buffer may remain outstanding -> overlap).
            // s_wait_tensorcnt takes a constant immediate -> branch on literal.
            if (issuer) {
                if (it + 1 < nt) __builtin_amdgcn_s_wait_tensorcnt(2);
                else             __builtin_amdgcn_s_wait_tensorcnt(0);
            }
#endif
            __syncthreads();
            if (active) {
                const __bf16* a = &sA[cur][(w * 16 + l15) * KT];
                const __bf16* b = &sB[cur][l15 * KT];
#pragma unroll
                for (int kk = 0; kk < KT; kk += 32) {
                    FragBF fa, fb;
                    fa.q[0] = *(const u32x4*)(a + kk + ho);
                    fa.q[1] = *(const u32x4*)(a + kk + 16 + ho);
                    fb.q[0] = *(const u32x4*)(b + kk + ho);
                    fb.q[1] = *(const u32x4*)(b + kk + 16 + ho);
                    acc = __builtin_amdgcn_wmma_f32_16x16x32_bf16(
                              false, fa.v, false, fb.v, (short)0, acc, false, false);
                }
            }
            __syncthreads();
        }
    }

    if (!active) return;
    // D-matrix lane map: elem r -> row = tm*16 + half*8 + r, col = tn*16 + l15
    int   ncol  = tn * 16 + l15;
    float bc    = bias[ncol];
    int   rbase = tm * 16 + half * 8;
    if (mode == 0) {
#pragma unroll
        for (int r = 0; r < 8; ++r)
            C[(long)(rbase + r) * ldc + ncol] = acc[r] + bc;
    } else {
#pragma unroll
        for (int r = 0; r < 8; ++r) {
            int   row = rbase + r;                       // batch index
            float sv  = acc[r] + bc + sx_t[(long)row * ld_sx + ncol];
            float g   = 1.f / (1.f + __expf(-sv));       // i=f=o gate
            long  ci  = (long)row * (long)N + ncol;
            float cn  = g * (cstate[ci] + tanhf(sv));    // g*c + g*tanh(s)
            cstate[ci] = cn;
            float hn  = g * tanhf(cn);
            hstate[ci] = hn;
            ys_t[(long)row * ld_ys + ncol] = hn;
        }
    }
}

// ------------------- fp32 [K,N] -> bf16 transposed [N,K] -------------------
__global__ __launch_bounds__(256) void wconv(const float* __restrict__ W,
                                             __bf16* __restrict__ Wt, int K, int N) {
    long i = (long)blockIdx.x * 256 + threadIdx.x;
    if (i >= (long)K * N) return;
    int k = (int)(i / N), n = (int)(i % N);
    Wt[(long)n * K + k] = (__bf16)W[i];
}

__global__ __launch_bounds__(256) void fzero(float* p, long n) {
    long i = (long)blockIdx.x * 256 + threadIdx.x;
    if (i < n) p[i] = 0.f;
}
__global__ __launch_bounds__(256) void fcopy(const float* __restrict__ s,
                                             float* __restrict__ d, long n) {
    long i = (long)blockIdx.x * 256 + threadIdx.x;
    if (i < n) d[i] = s[i];
}

// ---------------------------------------------------------------------------
extern "C" void kernel_launch(void* const* d_in, const int* in_sizes, int n_in,
                              void* d_out, int out_size, void* d_ws, size_t ws_size,
                              hipStream_t stream) {
    (void)in_sizes; (void)n_in; (void)out_size; (void)ws_size;
    constexpr int B = 64, S = 256, DIN = 256, KH = 64, H = 512, DOUT = 128;
    constexpr int T = B * S;
    constexpr int CT = 2048;           // token chunk (bounds scratch to ~87 MB)

    const float* x     = (const float*)d_in[0];
    const float* xA_ws = (const float*)d_in[1];
    const float* xA_wb = (const float*)d_in[2];
    const float* xA_bb = (const float*)d_in[3];
    const float* xB_ws = (const float*)d_in[4];
    const float* xB_wb = (const float*)d_in[5];
    const float* xB_bb = (const float*)d_in[6];
    const float* hA_ws = (const float*)d_in[7];
    const float* hA_wb = (const float*)d_in[8];
    const float* hA_bb = (const float*)d_in[9];
    const float* hB_ws = (const float*)d_in[10];
    const float* hB_wb = (const float*)d_in[11];
    const float* hB_bb = (const float*)d_in[12];
    const float* fc_ws = (const float*)d_in[13];
    const float* fc_wb = (const float*)d_in[14];
    const float* fc_bb = (const float*)d_in[15];
    float* outp = (float*)d_out;

    char*  ws  = (char*)d_ws;
    size_t off = 0;
    auto alloc = [&](size_t bytes) -> char* {
        char* p = ws + off;
        off = (off + bytes + 255) & ~(size_t)255;
        return p;
    };
    __bf16* wt_xAws = (__bf16*)alloc((size_t)1280 * 64 * 2);
    __bf16* wt_xAwb = (__bf16*)alloc((size_t)256  * 64 * 2);
    __bf16* wt_xBws = (__bf16*)alloc((size_t)320  * 512 * 2);
    __bf16* wt_xBwb = (__bf16*)alloc((size_t)64   * 512 * 2);
    __bf16* wt_hAws = (__bf16*)alloc((size_t)2560 * 64 * 2);
    __bf16* wt_hAwb = (__bf16*)alloc((size_t)512  * 64 * 2);
    __bf16* wt_hBws = (__bf16*)alloc((size_t)320  * 512 * 2);
    __bf16* wt_hBwb = (__bf16*)alloc((size_t)64   * 512 * 2);
    __bf16* wt_fcws = (__bf16*)alloc((size_t)2560 * 128 * 2);
    __bf16* wt_fcwb = (__bf16*)alloc((size_t)512  * 128 * 2);
    __bf16* basis  = (__bf16*)alloc((size_t)CT * 2560 * 2);
    __bf16* silu   = (__bf16*)alloc((size_t)CT * 512  * 2);
    float*  u1     = (float*) alloc((size_t)T  * 64   * 4);
    float*  sx     = (float*) alloc((size_t)T  * 512  * 4);
    float*  ys     = (float*) alloc((size_t)T  * 512  * 4);
    float*  hst    = (float*) alloc((size_t)B  * H * 4);
    float*  cst    = (float*) alloc((size_t)B  * H * 4);
    __bf16* basisH = (__bf16*)alloc((size_t)B * 2560 * 2);
    __bf16* siluH  = (__bf16*)alloc((size_t)B * 512  * 2);
    float*  ub     = (float*) alloc((size_t)B * 64 * 4);
    __bf16* basisU = (__bf16*)alloc((size_t)B * 320 * 2);
    __bf16* siluU  = (__bf16*)alloc((size_t)B * 64  * 2);

    auto conv = [&](const float* W, __bf16* Wt, int K, int N) {
        long n = (long)K * N;
        wconv<<<(int)((n + 255) / 256), 256, 0, stream>>>(W, Wt, K, N);
    };
    auto prep = [&](const float* X, long ldx, int rows, int D, __bf16* b, __bf16* s) {
        kan_prep<<<(rows + 7) / 8, 256, 0, stream>>>(X, ldx, rows, D, b, s);
    };
    auto gemm = [&](const __bf16* A1, const __bf16* W1, int K1,
                    const __bf16* A2, const __bf16* W2, int K2,
                    const float* bias, float* C, long ldc, int M, int N,
                    int mode, const float* sxt, long ldsx,
                    float* cs, float* hs, float* yst, long ldys) {
        int mblocks = ((M / 16) + 7) / 8;      // 128-row strips
        int grid = mblocks * (N / 16);
        kan_gemm<<<grid, 256, 0, stream>>>(
            A1, W1, K1, A2, W2, K2, bias, C, ldc, M, N, mode,
            sxt, ldsx, cs, hs, yst, ldys);
    };

    // -------- phase 0: weight convert/transpose to bf16 --------
    conv(xA_ws, wt_xAws, 1280, 64);
    conv(xA_wb, wt_xAwb, 256,  64);
    conv(xB_ws, wt_xBws, 320,  512);
    conv(xB_wb, wt_xBwb, 64,   512);
    conv(hA_ws, wt_hAws, 2560, 64);
    conv(hA_wb, wt_hAwb, 512,  64);
    conv(hB_ws, wt_hBws, 320,  512);
    conv(hB_wb, wt_hBwb, 64,   512);
    conv(fc_ws, wt_fcws, 2560, 128);
    conv(fc_wb, wt_fcwb, 512,  128);

    // -------- phase 1: u1 = kan_layer_A(x) ---------------------------------
    for (int ch = 0; ch < T / CT; ++ch) {
        prep(x + (long)ch * CT * DIN, DIN, CT, DIN, basis, silu);
        gemm(basis, wt_xAws, 1280, silu, wt_xAwb, 256, xA_bb,
             u1 + (long)ch * CT * KH, KH, CT, KH, 0,
             nullptr, 0, nullptr, nullptr, nullptr, 0);
    }
    // -------- phase 1b: sx = kan_layer_B(u1) -------------------------------
    for (int ch = 0; ch < T / CT; ++ch) {
        prep(u1 + (long)ch * CT * KH, KH, CT, KH, basis, silu);
        gemm(basis, wt_xBws, 320, silu, wt_xBwb, 64, xB_bb,
             sx + (long)ch * CT * H, H, CT, H, 0,
             nullptr, 0, nullptr, nullptr, nullptr, 0);
    }

    // -------- phase 2: sequential LSTM scan --------------------------------
    fzero<<<(B * H + 255) / 256, 256, 0, stream>>>(hst, (long)B * H);
    fzero<<<(B * H + 255) / 256, 256, 0, stream>>>(cst, (long)B * H);
    for (int t = 0; t < S; ++t) {
        prep(hst, H, B, H, basisH, siluH);
        gemm(basisH, wt_hAws, 2560, siluH, wt_hAwb, 512, hA_bb,
             ub, KH, B, KH, 0, nullptr, 0, nullptr, nullptr, nullptr, 0);
        prep(ub, KH, B, KH, basisU, siluU);
        gemm(basisU, wt_hBws, 320, siluU, wt_hBwb, 64, hB_bb,
             hst /*unused*/, H, B, H, /*mode=*/1,
             sx + (long)t * H, (long)S * H,
             cst, hst, ys + (long)t * H, (long)S * H);
    }

    // -------- phase 3: out = kan_layer(ys, fc) -----------------------------
    for (int ch = 0; ch < T / CT; ++ch) {
        prep(ys + (long)ch * CT * H, H, CT, H, basis, silu);
        gemm(basis, wt_fcws, 2560, silu, wt_fcwb, 512, fc_bb,
             outp + (long)ch * CT * DOUT, DOUT, CT, DOUT, 0,
             nullptr, 0, nullptr, nullptr, nullptr, 0);
    }
    // tuple tail: (h_f[1,64,512], c_f[1,64,512])
    fcopy<<<(B * H + 255) / 256, 256, 0, stream>>>(hst, outp + (long)T * DOUT, (long)B * H);
    fcopy<<<(B * H + 255) / 256, 256, 0, stream>>>(cst, outp + (long)T * DOUT + (long)B * H, (long)B * H);
}
